// Attention_10857677324470
// MI455X (gfx1250) — matrix operand order repaired and checked by reading, not compiled, not run on hardware
//
#include <hip/hip_runtime.h>

// ---------------------------------------------------------------------------
// Focused linear attention for MI455X (gfx1250, wave32, WMMA bf16)
// Fixed problem sizes from the reference setup_inputs():
//   x: (4, 512, 128, 128) fp32, P=2, B=2, HEADS=8, Ch=64, N=16384
// ---------------------------------------------------------------------------

typedef __attribute__((ext_vector_type(16))) __bf16          v16bf;
typedef __attribute__((ext_vector_type(8)))  float           v8f;
typedef __attribute__((ext_vector_type(8)))  unsigned short  ushort8;

#define DEVFN __device__ __forceinline__

constexpr int NPIX = 16384;   // H*W
constexpr int KC   = 512;     // channels per path

DEVFN unsigned short f2bf(float f) {
  // native fptrunc -> v_cvt_(pk_)bf16_f32 on gfx1250
  union { __bf16 h; unsigned short s; } cv;
  cv.h = (__bf16)f;
  return cv.s;
}
DEVFN float bf2f(unsigned short s) {
  return __uint_as_float(((unsigned int)s) << 16);
}
DEVFN float sigmoidf(float x) { return 1.f / (1.f + __expf(-x)); }

union FragBF {
  v16bf v;
  unsigned short u[16];
  ushort8 h[2];
};

DEVFN v8f wmma_bf16(const FragBF& a, const FragBF& b, v8f c) {
  // v_wmma_f32_16x16x32_bf16  (neg_a, A, neg_b, B, c_mod, C, reuse_a, reuse_b)
  return __builtin_amdgcn_wmma_f32_16x16x32_bf16(false, a.v, false, b.v,
                                                 (short)0, c, false, false);
}

DEVFN v8f v8f_zero() { v8f z; for (int i = 0; i < 8; ++i) z[i] = 0.f; return z; }

// ---------------------------------------------------------------------------
// Tiled bf16 GEMM:  Out[pb][M][N] = W[path][M][K] * B[pb][K][N]
// Workgroup tile 128x128, 8 waves (4 along M x 2 along N), K step 32.
// A staged row-major in LDS, B staged transposed so every WMMA fragment is a
// pair of contiguous 16B (ds_load_b128) reads. Epilogue re-tiles each wave's
// 32x64 result through a private LDS slab so global stores are 128B lines.
// ---------------------------------------------------------------------------
template <bool B_BF16, bool OUT_BF16>
__global__ __launch_bounds__(256) void gemm128(const float* __restrict__ Wq,
                                               const void*  __restrict__ Bsrc,
                                               void*        __restrict__ Out,
                                               int M) {
  // staging (20480B) and epilogue slabs (8 * 16*68*4 = 34816B) share LDS
  __shared__ __align__(16) char smemraw[34816];
  unsigned short* As = (unsigned short*)smemraw;   // [128][40]
  unsigned short* Bt = As + 128 * 40;              // [128][40] (n-major, k inner)

  const int pb    = blockIdx.z;
  const int path  = pb >> 1;
  const int nBase = blockIdx.x * 128;
  const int mBase = blockIdx.y * 128;

  const float* W = Wq + (size_t)(path * M + mBase) * KC;
  const float*          Bf = (const float*)Bsrc          + (size_t)pb * KC * NPIX;
  const unsigned short* Bh = (const unsigned short*)Bsrc + (size_t)pb * KC * NPIX;

  const int tid   = threadIdx.x;
  const int lane  = tid & 31;
  const int wid   = tid >> 5;
  const int waveM = wid & 3;    // 4 waves along M, 32 rows each
  const int waveN = wid >> 2;   // 2 waves along N, 64 cols each
  const int l15   = lane & 15;
  const int klo   = (lane < 16) ? 0 : 8;
  const int koff  = (lane < 16) ? 0 : 16;

  v8f acc[2][4];
  for (int i = 0; i < 2; ++i)
    for (int j = 0; j < 4; ++j) acc[i][j] = v8f_zero();

  for (int k0 = 0; k0 < KC; k0 += 32) {
    // prefetch next K-chunk (global_prefetch_b8; one line per thread per operand)
    if (k0 + 32 < KC) {
      int mA = tid >> 5, kA = tid & 31;
      __builtin_prefetch((const void*)&W[(size_t)mA * KC + k0 + 32 + kA], 0, 3);
      int kB = tid >> 7, nB = tid & 127;
      if (B_BF16)
        __builtin_prefetch((const void*)&Bh[(size_t)(k0 + 32 + kB) * NPIX + nBase + nB], 0, 3);
      else
        __builtin_prefetch((const void*)&Bf[(size_t)(k0 + 32 + kB) * NPIX + nBase + nB], 0, 3);
    }
    // stage A: 128x32 fp32 -> bf16
    for (int j = 0; j < 16; ++j) {
      int id = tid + 256 * j;
      int m = id >> 5, k = id & 31;
      As[m * 40 + k] = f2bf(W[(size_t)m * KC + k0 + k]);
    }
    // stage B transposed: [k][n] global -> Bt[n][k]
    for (int j = 0; j < 16; ++j) {
      int id = tid + 256 * j;
      int k = id >> 7, n = id & 127;
      float val;
      if (B_BF16) val = bf2f(Bh[(size_t)(k0 + k) * NPIX + nBase + n]);
      else        val = Bf[(size_t)(k0 + k) * NPIX + nBase + n];
      Bt[n * 40 + k] = f2bf(val);
    }
    __syncthreads();

    FragBF afrag[2], bfrag[4];
    for (int mt = 0; mt < 2; ++mt) {
      const unsigned short* ap = &As[(waveM * 32 + mt * 16 + l15) * 40];
      afrag[mt].h[0] = *(const ushort8*)(ap + koff);       // K = koff..koff+7
      afrag[mt].h[1] = *(const ushort8*)(ap + koff + 8);   // K = koff+8..
    }
    for (int nt = 0; nt < 4; ++nt) {
      const unsigned short* bp = &Bt[(waveN * 64 + nt * 16 + l15) * 40 + koff];
      bfrag[nt].h[0] = *(const ushort8*)(bp);              // K = koff..koff+7
      bfrag[nt].h[1] = *(const ushort8*)(bp + 8);          // K = koff+8..
    }
    for (int mt = 0; mt < 2; ++mt)
      for (int nt = 0; nt < 4; ++nt)
        acc[mt][nt] = wmma_bf16(afrag[mt], bfrag[nt], acc[mt][nt]);
    __syncthreads();
  }

  // epilogue: per-wave LDS slab -> fully coalesced 128B global stores
  float* slab = (float*)smemraw + wid * (16 * 68);   // 16 rows x 64 cols, pad 68
  for (int mt = 0; mt < 2; ++mt) {
    for (int nt = 0; nt < 4; ++nt)
      for (int i = 0; i < 8; ++i) {
        int row = (lane < 16) ? i : 8 + i;             // D layout
        slab[row * 68 + nt * 16 + l15] = acc[mt][nt][i];
      }
    __syncthreads();
    if (OUT_BF16) {
      unsigned short* Oh = (unsigned short*)Out + (size_t)pb * M * NPIX;
      for (int it = 0; it < 16; ++it) {
        int row = it, col = lane * 2;
        unsigned int pk = (unsigned int)f2bf(slab[row * 68 + col]) |
                          ((unsigned int)f2bf(slab[row * 68 + col + 1]) << 16);
        int gm = mBase + waveM * 32 + mt * 16 + row;
        int gn = nBase + waveN * 64 + col;
        *(unsigned int*)(Oh + (size_t)gm * NPIX + gn) = pk;
      }
    } else {
      float* Of = (float*)Out + (size_t)pb * M * NPIX;
      for (int it = 0; it < 32; ++it) {
        int row = it >> 1, col = (it & 1) * 32 + lane;
        int gm = mBase + waveM * 32 + mt * 16 + row;
        int gn = nBase + waveN * 64 + col;
        Of[(size_t)gm * NPIX + gn] = slab[row * 68 + col];
      }
    }
    __syncthreads();
  }
}

// ---------------------------------------------------------------------------
// Fused depthwise 3x3 on qkv + per-pixel channel norms (q1,q2,k1,k2) + v.
//  q1/q2 written pixel-major [bh][n][64]; k1/k2/v channel-major [bh][64][n].
// ---------------------------------------------------------------------------
__global__ __launch_bounds__(128) void dw_norm(
    const unsigned short* __restrict__ qkv,  // [4][1536][16384] bf16
    const float* __restrict__ dww,           // [3072][9]
    unsigned short* __restrict__ q1o, unsigned short* __restrict__ q2o,
    unsigned short* __restrict__ k1o, unsigned short* __restrict__ k2o,
    unsigned short* __restrict__ vo) {
  extern __shared__ float smem[];            // qb[128][65] + kb[128][65]
  float* qb = smem;
  float* kb = smem + 128 * 65;

  const int bh = blockIdx.x;                 // pb*8 + h
  const int pb = bh >> 3, h = bh & 7;
  const int p  = pb >> 1;
  const int n  = blockIdx.y * 128 + threadIdx.x;
  const int y  = n >> 7, x = n & 127;
  const size_t base = (size_t)pb * 1536 * NPIX;

  float qn2 = 0.f, q2n2 = 0.f, kn2 = 0.f, k2n2 = 0.f;
  for (int ch = 0; ch < 64; ++ch) {
    int jq = h * 64 + ch;
    const float* wq = dww + (size_t)(p * 1536 + jq) * 9;
    const float* wk = dww + (size_t)(p * 1536 + 512 + jq) * 9;
    const float* wv = dww + (size_t)(p * 1536 + 1024 + jq) * 9;
    const unsigned short* pq = qkv + base + (size_t)jq * NPIX;
    const unsigned short* pk = qkv + base + (size_t)(512 + jq) * NPIX;
    const unsigned short* pv = qkv + base + (size_t)(1024 + jq) * NPIX;
    float sq = 0.f, sk = 0.f, sv = 0.f;
    for (int dy = -1; dy <= 1; ++dy) {
      int yy = y + dy;
      if (yy < 0 || yy > 127) continue;
      for (int dx = -1; dx <= 1; ++dx) {
        int xx = x + dx;
        if (xx < 0 || xx > 127) continue;
        int tap = (dy + 1) * 3 + (dx + 1);
        int off = yy * 128 + xx;
        sq += wq[tap] * bf2f(pq[off]);
        sk += wk[tap] * bf2f(pk[off]);
        sv += wv[tap] * bf2f(pv[off]);
      }
    }
    qb[threadIdx.x * 65 + ch] = sq;
    kb[threadIdx.x * 65 + ch] = sk;
    vo[((size_t)bh * 64 + ch) * NPIX + n] = f2bf(sv);
    qn2 += sq * sq;
    float qr = fmaxf(sq, 0.f); qr *= qr; qr *= qr;   // relu^4
    q2n2 += qr * qr;
    kn2 += sk * sk;
    float kr = fmaxf(sk, 0.f); kr *= kr; kr *= kr;
    k2n2 += kr * kr;
  }
  const float rq1 = 1.f / (sqrtf(qn2)  + 1e-8f);
  const float rq2 = 1.f / (sqrtf(q2n2) + 1e-8f);
  const float rk1 = 1.f / (sqrtf(kn2)  + 1e-8f);
  const float rk2 = 1.f / (sqrtf(k2n2) + 1e-8f);

  unsigned short* q1row = q1o + ((size_t)bh * NPIX + n) * 64;
  unsigned short* q2row = q2o + ((size_t)bh * NPIX + n) * 64;
  for (int ch = 0; ch < 64; ++ch) {
    float qv = qb[threadIdx.x * 65 + ch];
    float kv = kb[threadIdx.x * 65 + ch];
    float qr = fmaxf(qv, 0.f); qr *= qr; qr *= qr;
    float kr = fmaxf(kv, 0.f); kr *= kr; kr *= kr;
    q1row[ch] = f2bf(qv * rq1);
    q2row[ch] = f2bf(qr * rq2);
    k1o[((size_t)bh * 64 + ch) * NPIX + n] = f2bf(kv * rk1);
    k2o[((size_t)bh * 64 + ch) * NPIX + n] = f2bf(kr * rk2);
  }
}

// ---------------------------------------------------------------------------
__global__ void zero_f32(float* p, int nelem) {
  int i = blockIdx.x * 256 + threadIdx.x;
  if (i < nelem) p[i] = 0.f;
}

// column sums over n: vsum, k1sum, k2sum (k2 pre-scaled by sigmoid(scale))
__global__ __launch_bounds__(256) void reduce_sums(
    const unsigned short* __restrict__ v, const unsigned short* __restrict__ k1,
    const unsigned short* __restrict__ k2, const float* __restrict__ scale,
    float* __restrict__ vsum, float* __restrict__ k1sum,
    float* __restrict__ k2sums) {
  __shared__ float red[256];
  const int row = blockIdx.x;   // bh*64 + c
  const int arr = blockIdx.y;   // 0:v 1:k1 2:k2
  const unsigned short* src = (arr == 0) ? v : (arr == 1) ? k1 : k2;
  float s = 0.f;
  for (int n = threadIdx.x; n < NPIX; n += 256)
    s += bf2f(src[(size_t)row * NPIX + n]);
  red[threadIdx.x] = s;
  __syncthreads();
  for (int st = 128; st > 0; st >>= 1) {
    if (threadIdx.x < st) red[threadIdx.x] += red[threadIdx.x + st];
    __syncthreads();
  }
  if (threadIdx.x == 0) {
    float r = red[0];
    if (arr == 2) {
      int bh = row >> 6, p = (bh >> 3) >> 1, h = bh & 7;
      r *= sigmoidf(scale[p * 8 + h]);
    }
    float* dst = (arr == 0) ? vsum : (arr == 1) ? k1sum : k2sums;
    dst[row] = r;
  }
}

// ---------------------------------------------------------------------------
// attn1/attn2 = k{1,2} (64 x N)  x  v^T (N x 64), split-K with fp32 atomics.
// k and v are channel-major so both fragments are contiguous global b128 loads.
// ---------------------------------------------------------------------------
__global__ __launch_bounds__(256) void attn_mat(
    const unsigned short* __restrict__ k1, const unsigned short* __restrict__ k2,
    const unsigned short* __restrict__ v, float* __restrict__ a1f,
    float* __restrict__ a2f) {
  const int bh    = blockIdx.x;
  const int split = blockIdx.y;               // 16 K-splits of 1024
  const int lane  = threadIdx.x & 31;
  const int wid   = threadIdx.x >> 5;
  const int c0    = (wid & 3) * 16;           // A row tile
  const int d0    = (wid >> 2) * 32;          // two B col tiles
  const int l15   = lane & 15;
  const int klo   = (lane < 16) ? 0 : 8;
  const int koff  = (lane < 16) ? 0 : 16;

  const unsigned short* k1b = k1 + (size_t)bh * 64 * NPIX;
  const unsigned short* k2b = k2 + (size_t)bh * 64 * NPIX;
  const unsigned short* vb  = v  + (size_t)bh * 64 * NPIX;

  v8f acc1[2] = {v8f_zero(), v8f_zero()};
  v8f acc2[2] = {v8f_zero(), v8f_zero()};

  for (int ci = 0; ci < 32; ++ci) {
    int n0 = (split * 32 + ci) * 32;
    FragBF a1, a2, b[2];
    const unsigned short* r1 = k1b + (size_t)(c0 + l15) * NPIX + n0;
    a1.h[0] = *(const ushort8*)(r1 + koff);
    a1.h[1] = *(const ushort8*)(r1 + koff + 8);
    const unsigned short* r2 = k2b + (size_t)(c0 + l15) * NPIX + n0;
    a2.h[0] = *(const ushort8*)(r2 + koff);
    a2.h[1] = *(const ushort8*)(r2 + koff + 8);
    for (int dt = 0; dt < 2; ++dt) {
      const unsigned short* rv =
          vb + (size_t)(d0 + dt * 16 + l15) * NPIX + n0 + koff;
      b[dt].h[0] = *(const ushort8*)(rv);
      b[dt].h[1] = *(const ushort8*)(rv + 8);
    }
    for (int dt = 0; dt < 2; ++dt) {
      acc1[dt] = wmma_bf16(a1, b[dt], acc1[dt]);
      acc2[dt] = wmma_bf16(a2, b[dt], acc2[dt]);
    }
  }
  for (int dt = 0; dt < 2; ++dt)
    for (int i = 0; i < 8; ++i) {
      int cc = c0 + ((lane < 16) ? i : 8 + i);
      int dd = d0 + dt * 16 + l15;
      atomicAdd(&a1f[(size_t)bh * 4096 + cc * 64 + dd], acc1[dt][i]);
      atomicAdd(&a2f[(size_t)bh * 4096 + cc * 64 + dd], acc2[dt][i]);
    }
}

// transpose to [d][c] bf16; fold sigmoid(scale) into attn2
__global__ __launch_bounds__(256) void attn_finalize(
    const float* __restrict__ a1f, const float* __restrict__ a2f,
    const float* __restrict__ scale, unsigned short* __restrict__ a1T,
    unsigned short* __restrict__ a2T) {
  const int bh = blockIdx.x;
  const int p = (bh >> 3) >> 1, h = bh & 7;
  const float sg = sigmoidf(scale[p * 8 + h]);
  for (int id = threadIdx.x; id < 4096; id += 256) {
    int c = id >> 6, d = id & 63;
    a1T[(size_t)bh * 4096 + d * 64 + c] = f2bf(a1f[(size_t)bh * 4096 + id]);
    a2T[(size_t)bh * 4096 + d * 64 + c] = f2bf(a2f[(size_t)bh * 4096 + id] * sg);
  }
}

// ---------------------------------------------------------------------------
// CRPE: per-head grouped conv (64 ch -> 1) with window 3/5/7; sigmoid output.
// ---------------------------------------------------------------------------
__global__ __launch_bounds__(256) void crpe_kernel(
    const unsigned short* __restrict__ v,  // [32][64][16384]
    const float* __restrict__ w0, const float* __restrict__ b0,
    const float* __restrict__ w1, const float* __restrict__ b1,
    const float* __restrict__ w2, const float* __restrict__ b2,
    float* __restrict__ refine) {          // [32][16384]
  __shared__ float tile[22 * 22];
  const int t  = threadIdx.x;
  const int tx = t & 15, ty = t >> 4;
  const int x0 = (blockIdx.x & 7) * 16, y0 = (blockIdx.x >> 3) * 16;
  const int h  = blockIdx.y;
  const int pb = blockIdx.z;
  const int p  = pb >> 1;
  int win;
  const float* wgt;
  float bias;
  if (h < 2)      { win = 3; wgt = w0 + (size_t)(p * 2 + h) * 64 * 9;        bias = b0[p * 2 + h]; }
  else if (h < 5) { win = 5; wgt = w1 + (size_t)(p * 3 + (h - 2)) * 64 * 25; bias = b1[p * 3 + h - 2]; }
  else            { win = 7; wgt = w2 + (size_t)(p * 3 + (h - 5)) * 64 * 49; bias = b2[p * 3 + h - 5]; }
  const int r  = win >> 1;
  const int tw = 16 + 2 * r;
  const int bh = pb * 8 + h;

  float acc = 0.f;
  for (int ch = 0; ch < 64; ++ch) {
    const unsigned short* plane = v + ((size_t)bh * 64 + ch) * NPIX;
    __syncthreads();
    for (int id = t; id < tw * tw; id += 256) {
      int lx = id % tw, ly = id / tw;
      int gx = x0 + lx - r, gy = y0 + ly - r;
      float val = 0.f;
      if (gx >= 0 && gx < 128 && gy >= 0 && gy < 128)
        val = bf2f(plane[gy * 128 + gx]);
      tile[id] = val;
    }
    __syncthreads();
    const float* wc = wgt + ch * win * win;
    for (int dy = 0; dy < win; ++dy)
      for (int dx = 0; dx < win; ++dx)
        acc += wc[dy * win + dx] * tile[(ty + dy) * tw + (tx + dx)];
  }
  int n = (y0 + ty) * 128 + (x0 + tx);
  refine[(size_t)bh * NPIX + n] = sigmoidf(acc + bias);
}

// ---------------------------------------------------------------------------
// numer = vsum + q1@attn1 + q2@attn2'  via chained WMMAs (C = vsum broadcast);
// out = numer * (temp/denom) + refine; LDS re-tile for coalesced bf16 stores.
// ---------------------------------------------------------------------------
__global__ __launch_bounds__(256) void attn_apply(
    const unsigned short* __restrict__ q1, const unsigned short* __restrict__ q2,
    const unsigned short* __restrict__ a1T, const unsigned short* __restrict__ a2T,
    const float* __restrict__ vsum, const float* __restrict__ k1sum,
    const float* __restrict__ k2sums, const float* __restrict__ refine,
    const float* __restrict__ temperature,
    unsigned short* __restrict__ attn_out) {   // [32*64][16384]
  __shared__ unsigned short ot[64 * 136];
  __shared__ float dscale[128];
  __shared__ float radd[128];

  const int bh = blockIdx.x;
  const int nt = blockIdx.y;
  const int p = (bh >> 3) >> 1, h = bh & 7;
  const float temp = temperature[p * 8 + h];
  const int t = threadIdx.x;
  const unsigned short* q1b = q1 + (size_t)bh * NPIX * 64;
  const unsigned short* q2b = q2 + (size_t)bh * NPIX * 64;

  if (t < 128) {
    int n = nt * 128 + t;
    const unsigned short* r1 = q1b + (size_t)n * 64;
    const unsigned short* r2 = q2b + (size_t)n * 64;
    float s1 = 0.f, s2 = 0.f;
    for (int c2 = 0; c2 < 64; ++c2) {
      s1 += bf2f(r1[c2]) * k1sum[bh * 64 + c2];
      s2 += bf2f(r2[c2]) * k2sums[bh * 64 + c2];
    }
    float denom = 16384.f + s1 + s2 + 1e-8f;
    dscale[t] = temp / denom;
    radd[t]   = refine[(size_t)bh * NPIX + n];
  }
  __syncthreads();

  const int lane = t & 31, wid = t >> 5;
  const int l15  = lane & 15;
  const int klo  = (lane < 16) ? 0 : 8;
  const int koff = (lane < 16) ? 0 : 16;
  const int nSub = wid * 16;

  v8f acc[4];
  for (int dt = 0; dt < 4; ++dt) {
    float vs = vsum[bh * 64 + dt * 16 + l15];
    for (int i = 0; i < 8; ++i) acc[dt][i] = vs;
  }
  const unsigned short* a1b = a1T + (size_t)bh * 4096;
  const unsigned short* a2b = a2T + (size_t)bh * 4096;

  for (int kc = 0; kc < 64; kc += 32) {
    FragBF aq1, aq2;
    int n = nt * 128 + nSub + l15;
    const unsigned short* r1 = q1b + (size_t)n * 64 + kc;
    aq1.h[0] = *(const ushort8*)(r1 + koff);
    aq1.h[1] = *(const ushort8*)(r1 + koff + 8);
    const unsigned short* r2 = q2b + (size_t)n * 64 + kc;
    aq2.h[0] = *(const ushort8*)(r2 + koff);
    aq2.h[1] = *(const ushort8*)(r2 + koff + 8);
    for (int dt = 0; dt < 4; ++dt) {
      FragBF b1, b2;
      const unsigned short* p1 = a1b + (size_t)(dt * 16 + l15) * 64 + kc + koff;
      b1.h[0] = *(const ushort8*)(p1);
      b1.h[1] = *(const ushort8*)(p1 + 8);
      const unsigned short* p2 = a2b + (size_t)(dt * 16 + l15) * 64 + kc + koff;
      b2.h[0] = *(const ushort8*)(p2);
      b2.h[1] = *(const ushort8*)(p2 + 8);
      acc[dt] = wmma_bf16(aq1, b1, acc[dt]);
      acc[dt] = wmma_bf16(aq2, b2, acc[dt]);
    }
  }

  for (int dt = 0; dt < 4; ++dt) {
    int d = dt * 16 + l15;
    for (int i = 0; i < 8; ++i) {
      int nr = nSub + ((lane < 16) ? i : 8 + i);
      float val = acc[dt][i] * dscale[nr] + radd[nr];
      ot[d * 136 + nr] = f2bf(val);
    }
  }
  __syncthreads();
  for (int id = t; id < 64 * 128; id += 256) {
    int d = id >> 7, nn = id & 127;
    attn_out[((size_t)bh * 64 + d) * NPIX + nt * 128 + nn] = ot[d * 136 + nn];
  }
}

// ---------------------------------------------------------------------------
extern "C" void kernel_launch(void* const* d_in, const int* in_sizes, int n_in,
                              void* d_out, int out_size, void* d_ws,
                              size_t ws_size, hipStream_t stream) {
  (void)in_sizes; (void)n_in; (void)out_size; (void)ws_size;
  const float* x      = (const float*)d_in[0];
  const float* qkv_w  = (const float*)d_in[1];
  const float* dw_w   = (const float*)d_in[2];
  const float* proj_w = (const float*)d_in[3];
  const float* temper = (const float*)d_in[4];
  const float* scale  = (const float*)d_in[5];
  const float* cw0 = (const float*)d_in[6];  const float* cb0 = (const float*)d_in[7];
  const float* cw1 = (const float*)d_in[8];  const float* cb1 = (const float*)d_in[9];
  const float* cw2 = (const float*)d_in[10]; const float* cb2 = (const float*)d_in[11];

  char* ws = (char*)d_ws;
  size_t off = 0;
  auto alloc = [&](size_t bytes) -> char* {
    char* p = ws + off;
    off += (bytes + 255) & ~(size_t)255;
    return p;
  };
  unsigned short* qkv_pre = (unsigned short*)alloc((size_t)4 * 1536 * NPIX * 2);
  unsigned short* q1      = (unsigned short*)alloc((size_t)32 * NPIX * 64 * 2);
  unsigned short* q2      = (unsigned short*)alloc((size_t)32 * NPIX * 64 * 2);
  unsigned short* k1      = (unsigned short*)alloc((size_t)32 * 64 * NPIX * 2);
  unsigned short* k2      = (unsigned short*)alloc((size_t)32 * 64 * NPIX * 2);
  unsigned short* vv      = (unsigned short*)alloc((size_t)32 * 64 * NPIX * 2);
  unsigned short* attn_o  = (unsigned short*)alloc((size_t)32 * 64 * NPIX * 2);
  float* a1f    = (float*)alloc((size_t)32 * 4096 * 4 * 2);   // a1f and a2f adjacent
  float* a2f    = a1f + (size_t)32 * 4096;
  unsigned short* a1T = (unsigned short*)alloc((size_t)32 * 4096 * 2 * 2);
  unsigned short* a2T = a1T + (size_t)32 * 4096;
  float* vsum   = (float*)alloc((size_t)3 * 32 * 64 * 4);
  float* k1sum  = vsum + 32 * 64;
  float* k2sums = vsum + 2 * 32 * 64;
  float* refine = (float*)alloc((size_t)32 * NPIX * 4);

  // 1) qkv 1x1 grouped conv as 4 GEMMs: 1536 x 16384 x 512 (fp32 in, bf16 out)
  gemm128<false, true><<<dim3(128, 12, 4), 256, 0, stream>>>(
      qkv_w, (const void*)x, (void*)qkv_pre, 1536);

  // 2) depthwise 3x3 + per-pixel norms
  dw_norm<<<dim3(32, 128), 128, 2 * 128 * 65 * sizeof(float), stream>>>(
      qkv_pre, dw_w, q1, q2, k1, k2, vv);

  // 3) zero the split-K accumulators (deterministic per call)
  zero_f32<<<(2 * 32 * 4096 + 255) / 256, 256, 0, stream>>>(a1f, 2 * 32 * 4096);

  // 4) column sums (vsum, k1sum, sigmoid(scale)*k2sum)
  reduce_sums<<<dim3(32 * 64, 3), 256, 0, stream>>>(vv, k1, k2, scale, vsum,
                                                    k1sum, k2sums);

  // 5) attn1/attn2 = k @ v^T (split-K WMMA + fp32 atomics)
  attn_mat<<<dim3(32, 16), 256, 0, stream>>>(k1, k2, vv, a1f, a2f);

  // 6) transpose + scale-fold to bf16
  attn_finalize<<<32, 256, 0, stream>>>(a1f, a2f, scale, a1T, a2T);

  // 7) CRPE refine weights (sigmoid)
  crpe_kernel<<<dim3(64, 8, 4), 256, 0, stream>>>(vv, cw0, cb0, cw1, cb1, cw2,
                                                  cb2, refine);

  // 8) numerator/denominator, temperature, refine -> attn_out bf16
  attn_apply<<<dim3(32, 128), 256, 0, stream>>>(q1, q2, a1T, a2T, vsum, k1sum,
                                                k2sums, refine, temper, attn_o);

  // 9) proj 1x1 grouped conv: 512 x 16384 x 512 (bf16 in, fp32 out)
  gemm128<true, false><<<dim3(128, 4, 4), 256, 0, stream>>>(
      proj_w, (const void*)attn_o, d_out, 512);
}
